// MultiHeadAttention_85091892068585
// MI455X (gfx1250) — matrix operand order repaired and checked by reading, not compiled, not run on hardware
//
#include <hip/hip_runtime.h>

// ---------------------------------------------------------------------------
// CDNA5 (gfx1250) multi-head attention block, bf16 WMMA pipeline.
// B=2, L=2048, D_MODEL=1024, H=16, DK=64.  d_out = [out (B*L*D) | attn (B*H*L*L)]
// Each wave computes a 32x64 C tile (2x4 WMMA accumulators, B frag reused 2x).
// ---------------------------------------------------------------------------

typedef __attribute__((ext_vector_type(16))) __bf16 v16bf;
typedef __attribute__((ext_vector_type(8)))  __bf16 v8bf;
typedef __attribute__((ext_vector_type(4)))  __bf16 v4bf;
typedef __attribute__((ext_vector_type(8)))  float  v8f;
typedef __attribute__((ext_vector_type(4)))  float  v4f;

#define D_MODEL 1024
#define NHEADS  16
#define DK      64
#define SEQL    2048
#define NB      2

// ---------------- fragment loaders (per ISA §7.12.2 VGPR layouts) ----------

// A fragment 16x32 bf16: lane holds row M=lane%16.
// lanes 0-15 : elems 0-7 = K 0-7,  elems 8-15 = K 16-23
// lanes16-31 : elems 0-7 = K 8-15, elems 8-15 = K 24-31
__device__ inline v16bf load_a_frag(const __bf16* __restrict A, long lda, int lane) {
    int m  = lane & 15;
    int kb = (lane >> 4) << 3;               // 0 or 8
    const __bf16* p = A + (long)m * lda + kb;
    v8bf lo = *(const v8bf*)(p);
    v8bf hi = *(const v8bf*)(p + 16);
    v16bf r;
#pragma unroll
    for (int e = 0; e < 8; ++e) { r[e] = lo[e]; r[8 + e] = hi[e]; }
    return r;
}

// Same A fragment but sourced from f32 (converted to bf16 in-register).
__device__ inline v16bf load_a_frag_f32(const float* __restrict A, long lda, int lane) {
    int m  = lane & 15;
    int kb = (lane >> 4) << 3;
    const float* p = A + (long)m * lda + kb;
    v4f a0 = *(const v4f*)(p);
    v4f a1 = *(const v4f*)(p + 4);
    v4f a2 = *(const v4f*)(p + 16);
    v4f a3 = *(const v4f*)(p + 20);
    v16bf r;
#pragma unroll
    for (int e = 0; e < 4; ++e) {
        r[e]      = (__bf16)a0[e];
        r[4 + e]  = (__bf16)a1[e];
        r[8 + e]  = (__bf16)a2[e];
        r[12 + e] = (__bf16)a3[e];
    }
    return r;
}

// B fragment 32x16 bf16, operand stored as N x K row-major (computing A*B^T).
// lane holds column N=lane%16; elems 0-7 = K kb..kb+7, elems 8-15 = K kb+16..kb+23 (kb = lane<16 ? 0 : 8).
__device__ inline v16bf load_b_frag(const __bf16* __restrict Bm, long ldb, int lane) {
    int n  = lane & 15;
    int ko = (lane >> 4) << 3;               // 0 or 8
    const __bf16* p = Bm + (long)n * ldb + ko;
    v8bf lo = *(const v8bf*)(p);
    v8bf hi = *(const v8bf*)(p + 16);
    v16bf r;
#pragma unroll
    for (int e = 0; e < 8; ++e) { r[e] = lo[e]; r[8 + e] = hi[e]; }
    return r;
}

#define WMMA_BF16(A_, B_, C_) \
    __builtin_amdgcn_wmma_f32_16x16x32_bf16(false, (A_), false, (B_), (short)0, (C_), false, false)

// ---------------- elementwise cast -----------------------------------------

__global__ void cast_bf16_kernel(const float* __restrict src, __bf16* __restrict dst, long n4) {
    long i = ((long)blockIdx.x * blockDim.x + threadIdx.x);
    if (i < n4) {
        v4f v = *(const v4f*)(src + i * 4);
        v4bf o;
#pragma unroll
        for (int e = 0; e < 4; ++e) o[e] = (__bf16)v[e];
        *(v4bf*)(dst + i * 4) = o;
    }
}

// ---------------- generic GEMM: C = A * B^T + bias --------------------------
// A: [Z][Mb x K] bf16 row-major (contiguous).  B: [N x K] bf16 row-major.
// Block = 128 threads (4 waves); each wave computes 32 rows x 64 cols.
// Block covers 128 rows x 64 cols.  Output via strides as f32 or bf16.

__global__ void gemm_abt_kernel(const __bf16* __restrict A,
                                const __bf16* __restrict Bw,
                                const float* __restrict bias,
                                float* __restrict Cf, __bf16* __restrict Cb,
                                long csm, long csn, long czstride,
                                int Mb, int K) {
    int lane = threadIdx.x & 31;
    int wave = threadIdx.x >> 5;
    int m0 = blockIdx.x * 128 + wave * 32;
    int n0 = blockIdx.y * 64;
    int z  = blockIdx.z;

    const __bf16* Ab = A + ((long)z * Mb + m0) * (long)K;
    const __bf16* Bb = Bw + (long)n0 * (long)K;

    v8f acc[2][4];
#pragma unroll
    for (int s = 0; s < 2; ++s)
#pragma unroll
        for (int t = 0; t < 4; ++t) acc[s][t] = (v8f)(0.0f);

    for (int k0 = 0; k0 < K; k0 += 32) {
        if (k0 + 32 < K) {
            __builtin_prefetch(Ab + k0 + 32, 0, 1);
            __builtin_prefetch(Bb + k0 + 32, 0, 1);
        }
        v16bf a0 = load_a_frag(Ab + k0, K, lane);
        v16bf a1 = load_a_frag(Ab + (long)16 * K + k0, K, lane);
#pragma unroll
        for (int t = 0; t < 4; ++t) {
            v16bf b = load_b_frag(Bb + (long)(t * 16) * K + k0, K, lane);
            acc[0][t] = WMMA_BF16(a0, b, acc[0][t]);
            acc[1][t] = WMMA_BF16(a1, b, acc[1][t]);
        }
    }

#pragma unroll
    for (int s = 0; s < 2; ++s) {
#pragma unroll
        for (int t = 0; t < 4; ++t) {
            int n = n0 + t * 16 + (lane & 15);
            float bv = bias[n];
#pragma unroll
            for (int r = 0; r < 8; ++r) {
                int m = m0 + s * 16 + r + ((lane >> 4) << 3);
                long addr = (long)z * czstride + (long)m * csm + (long)n * csn;
                float val = acc[s][t][r] + bv;
                if (Cf) Cf[addr] = val;
                else    Cb[addr] = (__bf16)val;
            }
        }
    }
}

// ---------------- scores: S = (Qh * Kh^T) / 8  → f32 into attn region -------

__global__ void scores_kernel(const __bf16* __restrict Qb,
                              const __bf16* __restrict Kb,
                              float* __restrict attn) {
    int lane = threadIdx.x & 31;
    int wave = threadIdx.x >> 5;
    int m0 = blockIdx.x * 128 + wave * 32;
    int n0 = blockIdx.y * 64;
    int zz = blockIdx.z;            // b*16 + h
    int b  = zz >> 4;
    int h  = zz & 15;

    const __bf16* Ab = Qb + ((long)b * SEQL + m0) * D_MODEL + h * DK;
    const __bf16* Bb = Kb + ((long)b * SEQL + n0) * D_MODEL + h * DK;

    v8f acc[2][4];
#pragma unroll
    for (int s = 0; s < 2; ++s)
#pragma unroll
        for (int t = 0; t < 4; ++t) acc[s][t] = (v8f)(0.0f);

#pragma unroll
    for (int k0 = 0; k0 < DK; k0 += 32) {
        v16bf a0 = load_a_frag(Ab + k0, D_MODEL, lane);
        v16bf a1 = load_a_frag(Ab + (long)16 * D_MODEL + k0, D_MODEL, lane);
#pragma unroll
        for (int t = 0; t < 4; ++t) {
            v16bf bfrag = load_b_frag(Bb + (long)(t * 16) * D_MODEL + k0, D_MODEL, lane);
            acc[0][t] = WMMA_BF16(a0, bfrag, acc[0][t]);
            acc[1][t] = WMMA_BF16(a1, bfrag, acc[1][t]);
        }
    }

    float* arow = attn + (long)zz * SEQL * SEQL;
#pragma unroll
    for (int s = 0; s < 2; ++s) {
#pragma unroll
        for (int t = 0; t < 4; ++t) {
            int n = n0 + t * 16 + (lane & 15);
#pragma unroll
            for (int r = 0; r < 8; ++r) {
                int m = m0 + s * 16 + r + ((lane >> 4) << 3);
                arow[(long)m * SEQL + n] = acc[s][t][r] * 0.125f;   // 1/sqrt(64)
            }
        }
    }
}

// ---------------- softmax (with mask), in place on attn rows ----------------

__global__ void softmax_kernel(float* __restrict attn, const int* __restrict mask) {
    long r = blockIdx.x;                  // 0 .. B*H*L-1
    int q = (int)(r & (SEQL - 1));
    long bh = r >> 11;                    // / SEQL
    int b = (int)(bh >> 4);
    float* p = attn + r * (long)SEQL;
    const int* mrow = mask + (((long)b * SEQL) + q) * (long)SEQL;

    __shared__ float red[256];
    int t = threadIdx.x;
    float vals[8];
    float vmax = -3.4e38f;
#pragma unroll
    for (int j = 0; j < 8; ++j) {
        int idx = t + j * 256;
        float s = p[idx];
        if (mrow[idx] == 0) s = -1.0e9f;
        vals[j] = s;
        vmax = fmaxf(vmax, s);
    }
    red[t] = vmax;
    __syncthreads();
    for (int off = 128; off > 0; off >>= 1) {
        if (t < off) red[t] = fmaxf(red[t], red[t + off]);
        __syncthreads();
    }
    vmax = red[0];
    __syncthreads();

    float sum = 0.0f;
#pragma unroll
    for (int j = 0; j < 8; ++j) {
        vals[j] = __expf(vals[j] - vmax);
        sum += vals[j];
    }
    red[t] = sum;
    __syncthreads();
    for (int off = 128; off > 0; off >>= 1) {
        if (t < off) red[t] += red[t + off];
        __syncthreads();
    }
    float inv = 1.0f / red[0];
#pragma unroll
    for (int j = 0; j < 8; ++j) p[t + j * 256] = vals[j] * inv;
}

// ---------------- attn @ V : A (f32→bf16 on load) * Vt^T --------------------
// Vt stored [B][D_MODEL][L] bf16 (per-head transposed V), so this is A*B^T.

__global__ void av_kernel(const float* __restrict attn,
                          const __bf16* __restrict Vt,
                          __bf16* __restrict Ob) {
    int lane = threadIdx.x & 31;
    int wave = threadIdx.x >> 5;
    int m0 = blockIdx.x * 128 + wave * 32;
    int zz = blockIdx.z;
    int b  = zz >> 4;
    int h  = zz & 15;

    const float*  Ab = attn + ((long)zz * SEQL + m0) * (long)SEQL;
    const __bf16* Bb = Vt + ((long)b * D_MODEL + h * DK) * (long)SEQL;

    v8f acc[2][4];
#pragma unroll
    for (int s = 0; s < 2; ++s)
#pragma unroll
        for (int t = 0; t < 4; ++t) acc[s][t] = (v8f)(0.0f);

    for (int k0 = 0; k0 < SEQL; k0 += 32) {
        if (k0 + 32 < SEQL) {
            __builtin_prefetch(Ab + k0 + 32, 0, 1);
            __builtin_prefetch(Bb + k0 + 32, 0, 1);
        }
        v16bf a0 = load_a_frag_f32(Ab + k0, SEQL, lane);
        v16bf a1 = load_a_frag_f32(Ab + (long)16 * SEQL + k0, SEQL, lane);
#pragma unroll
        for (int t = 0; t < 4; ++t) {
            v16bf bfrag = load_b_frag(Bb + (long)(t * 16) * SEQL + k0, SEQL, lane);
            acc[0][t] = WMMA_BF16(a0, bfrag, acc[0][t]);
            acc[1][t] = WMMA_BF16(a1, bfrag, acc[1][t]);
        }
    }

#pragma unroll
    for (int s = 0; s < 2; ++s) {
#pragma unroll
        for (int t = 0; t < 4; ++t) {
            int n = t * 16 + (lane & 15);          // d within head
#pragma unroll
            for (int r = 0; r < 8; ++r) {
                int m = m0 + s * 16 + r + ((lane >> 4) << 3);
                Ob[((long)b * SEQL + m) * D_MODEL + h * DK + n] = (__bf16)acc[s][t][r];
            }
        }
    }
}

// ---------------- residual add + LayerNorm ----------------------------------

__global__ void ln_kernel(const float* __restrict obuf, const float* __restrict resid,
                          const float* __restrict g, const float* __restrict bta,
                          float* __restrict out) {
    long r = blockIdx.x;
    const float* xo = obuf + r * (long)D_MODEL;
    const float* xr = resid + r * (long)D_MODEL;
    int t = threadIdx.x;

    __shared__ float red[256];
    __shared__ float red2[256];

    float vals[4];
    float s = 0.0f, s2 = 0.0f;
#pragma unroll
    for (int j = 0; j < 4; ++j) {
        int idx = t + j * 256;
        float x = xo[idx] + xr[idx];
        vals[j] = x;
        s += x;
        s2 += x * x;
    }
    red[t] = s; red2[t] = s2;
    __syncthreads();
    for (int off = 128; off > 0; off >>= 1) {
        if (t < off) { red[t] += red[t + off]; red2[t] += red2[t + off]; }
        __syncthreads();
    }
    float mu  = red[0] * (1.0f / D_MODEL);
    float var = red2[0] * (1.0f / D_MODEL) - mu * mu;
    float inv = rsqrtf(var + 1e-5f);
    float* orow = out + r * (long)D_MODEL;
#pragma unroll
    for (int j = 0; j < 4; ++j) {
        int idx = t + j * 256;
        orow[idx] = (vals[j] - mu) * inv * g[idx] + bta[idx];
    }
}

// ---------------------------------------------------------------------------

extern "C" void kernel_launch(void* const* d_in, const int* in_sizes, int n_in,
                              void* d_out, int out_size, void* d_ws, size_t ws_size,
                              hipStream_t stream) {
    const float* q    = (const float*)d_in[0];
    const float* k    = (const float*)d_in[1];
    const float* v    = (const float*)d_in[2];
    const float* Wq   = (const float*)d_in[3];
    const float* bq   = (const float*)d_in[4];
    const float* Wk   = (const float*)d_in[5];
    const float* bk   = (const float*)d_in[6];
    const float* Wv   = (const float*)d_in[7];
    const float* bv   = (const float*)d_in[8];
    const float* Wo   = (const float*)d_in[9];
    const float* bo   = (const float*)d_in[10];
    const float* ln_g = (const float*)d_in[11];
    const float* ln_b = (const float*)d_in[12];
    const int*   mask = (const int*)d_in[13];

    const long M  = (long)NB * SEQL;          // 4096
    const long MD = M * D_MODEL;              // 4.19M
    const long WW = (long)D_MODEL * D_MODEL;  // 1.05M

    float* out  = (float*)d_out;
    float* attn = out + MD;                   // B*H*L*L f32

    char* p = (char*)d_ws;
    auto alloc = [&](size_t bytes) -> char* {
        char* r = p;
        p += (bytes + 255) & ~(size_t)255;
        return r;
    };
    __bf16* qbf  = (__bf16*)alloc(MD * 2);
    __bf16* kbf  = (__bf16*)alloc(MD * 2);
    __bf16* vbf  = (__bf16*)alloc(MD * 2);
    __bf16* Wqb  = (__bf16*)alloc(WW * 2);
    __bf16* Wkb  = (__bf16*)alloc(WW * 2);
    __bf16* Wvb  = (__bf16*)alloc(WW * 2);
    __bf16* Wob  = (__bf16*)alloc(WW * 2);
    __bf16* Qb   = (__bf16*)alloc(MD * 2);
    __bf16* Kb   = (__bf16*)alloc(MD * 2);
    __bf16* Vt   = (__bf16*)alloc(MD * 2);    // [B][D_MODEL][L]
    __bf16* Ob   = (__bf16*)alloc(MD * 2);    // attention output, bf16
    float*  obuf = (float*)alloc(MD * 4);     // out-projection, f32

    // casts (all sizes are multiples of 4)
    {
        long n4 = MD / 4;
        dim3 g((unsigned)((n4 + 255) / 256)), b(256);
        cast_bf16_kernel<<<g, b, 0, stream>>>(q, qbf, n4);
        cast_bf16_kernel<<<g, b, 0, stream>>>(k, kbf, n4);
        cast_bf16_kernel<<<g, b, 0, stream>>>(v, vbf, n4);
        long w4 = WW / 4;
        dim3 gw((unsigned)((w4 + 255) / 256));
        cast_bf16_kernel<<<gw, b, 0, stream>>>(Wq, Wqb, w4);
        cast_bf16_kernel<<<gw, b, 0, stream>>>(Wk, Wkb, w4);
        cast_bf16_kernel<<<gw, b, 0, stream>>>(Wv, Wvb, w4);
        cast_bf16_kernel<<<gw, b, 0, stream>>>(Wo, Wob, w4);
    }

    dim3 blk(128);                                   // 4 waves (wave32)
    dim3 gproj(SEQL / 128, D_MODEL / 64, NB);        // 16 x 16 x 2

    // Q/K projections → [B,L,D] bf16 ; V projection → transposed [B,D,L] bf16
    gemm_abt_kernel<<<gproj, blk, 0, stream>>>(qbf, Wqb, bq, nullptr, Qb,
                                               (long)D_MODEL, 1L, (long)SEQL * D_MODEL,
                                               SEQL, D_MODEL);
    gemm_abt_kernel<<<gproj, blk, 0, stream>>>(kbf, Wkb, bk, nullptr, Kb,
                                               (long)D_MODEL, 1L, (long)SEQL * D_MODEL,
                                               SEQL, D_MODEL);
    gemm_abt_kernel<<<gproj, blk, 0, stream>>>(vbf, Wvb, bv, nullptr, Vt,
                                               1L, (long)SEQL, (long)D_MODEL * SEQL,
                                               SEQL, D_MODEL);

    // scores → f32 directly into d_out attn region
    dim3 gsc(SEQL / 128, SEQL / 64, NB * NHEADS);    // 16 x 32 x 32
    scores_kernel<<<gsc, blk, 0, stream>>>(Qb, Kb, attn);

    // masked softmax in place
    softmax_kernel<<<dim3((unsigned)(NB * NHEADS * SEQL)), dim3(256), 0, stream>>>(attn, mask);

    // attn @ V  (attn f32 converted to bf16 on load)
    av_kernel<<<dim3(SEQL / 128, 1, NB * NHEADS), blk, 0, stream>>>(attn, Vt, Ob);

    // output projection → f32
    gemm_abt_kernel<<<gproj, blk, 0, stream>>>(Ob, Wob, bo, obuf, nullptr,
                                               (long)D_MODEL, 1L, (long)SEQL * D_MODEL,
                                               SEQL, D_MODEL);

    // residual + LayerNorm → d_out[0 .. M*D)
    ln_kernel<<<dim3((unsigned)M), dim3(256), 0, stream>>>(obuf, q, ln_g, ln_b, out);
}
